// CompressCausalSelfAttention_87591563035192
// MI455X (gfx1250) — matrix operand order repaired and checked by reading, not compiled, not run on hardware
//
#include <hip/hip_runtime.h>

typedef __attribute__((ext_vector_type(16))) __bf16 bf16x16;
typedef __attribute__((ext_vector_type(8)))  __bf16 bf16x8;
typedef __attribute__((ext_vector_type(8)))  float  f32x8;
typedef __attribute__((ext_vector_type(4)))  int    i32x4;

union V16U { bf16x16 v; bf16x8 h[2]; };

__device__ __forceinline__ bf16x16 load_a_op(const __bf16* p0, const __bf16* p1) {
    V16U u;
    u.h[0] = *(const bf16x8*)p0;   // 16B aligned
    u.h[1] = *(const bf16x8*)p1;
    return u.v;
}

__device__ __forceinline__ f32x8 wmma_bf16(bf16x16 a, bf16x16 b, f32x8 c) {
    // D = A(16x32 bf16) x B(32x16 bf16) + C(16x16 f32)
    return __builtin_amdgcn_wmma_f32_16x16x32_bf16(false, a, false, b, (short)0, c, false, false);
}

__device__ __forceinline__ float halfmax16(float v) {
    #pragma unroll
    for (int m = 1; m < 16; m <<= 1) v = fmaxf(v, __shfl_xor(v, m, 32));
    return v;
}
__device__ __forceinline__ float halfsum16(float v) {
    #pragma unroll
    for (int m = 1; m < 16; m <<= 1) v += __shfl_xor(v, m, 32);
    return v;
}

// ---------------------------------------------------------------------------
// CDNA5 async global->LDS copy (ASYNCcnt-tracked), guarded with fallbacks.
// Toolchain signature (from diagnostics): (v4i AS1*, v4i AS3*, imm off, imm cpol)
// ---------------------------------------------------------------------------
#if defined(__has_builtin)
#  if __has_builtin(__builtin_amdgcn_global_load_async_to_lds_b128)
#    define HAVE_ASYNC_LDS 1
#  endif
#  if __has_builtin(__builtin_amdgcn_s_wait_asynccnt)
#    define HAVE_WAIT_ASYNC 1
#  endif
#endif

__device__ __forceinline__ void copy16_g2l(__bf16* l, const __bf16* g) {
#ifdef HAVE_ASYNC_LDS
    __builtin_amdgcn_global_load_async_to_lds_b128(
        (__attribute__((address_space(1))) i32x4*)(unsigned long long)g,
        (__attribute__((address_space(3))) i32x4*)(unsigned int)(unsigned long long)l,
        0, 0);
#else
    *(bf16x8*)l = *(const bf16x8*)g;
#endif
}

#ifdef HAVE_ASYNC_LDS
#  ifdef HAVE_WAIT_ASYNC
#    define ASYNC_WAIT(n) __builtin_amdgcn_s_wait_asynccnt(n)
#  else
#    define ASYNC_WAIT(n) asm volatile("s_wait_asynccnt %0" :: "i"(n) : "memory")
#  endif
#else
#  define ASYNC_WAIT(n) do {} while (0)
#endif

// stage a 64x64 bf16 tile (global row stride gstride) into LDS (row stride 72,
// padded to kill bank aliasing).  128 threads x 4 chunks of 16B, coalesced.
__device__ __forceinline__ void stage_tile(__bf16* __restrict__ ldst,
                                           const __bf16* __restrict__ gsrc,
                                           long gstride, int tid) {
    #pragma unroll
    for (int c = 0; c < 4; ++c) {
        int chunk = c * 128 + tid;          // 0..511
        int row = chunk >> 3;
        int off = (chunk & 7) * 8;
        copy16_g2l(ldst + row * 72 + off, gsrc + (long)row * gstride + off);
    }
}

// ---------------------------------------------------------------------------
// Packing kernels (f32 -> bf16, layout shuffles)
// ---------------------------------------------------------------------------
__global__ void pack_f32_bf16(const float* __restrict__ src, __bf16* __restrict__ dst, long n) {
    long i = (long)blockIdx.x * 256 + threadIdx.x;
    if (i < n) dst[i] = (__bf16)src[i];
}

// kvinb[b][r][c], r<256 -> cmem, r<1280 -> mem, else x.  (B=2, rows 2304, cols 1024)
__global__ void pack_kvin(const float* __restrict__ x, const float* __restrict__ mem,
                          const float* __restrict__ cmem, __bf16* __restrict__ dst) {
    long i = (long)blockIdx.x * 256 + threadIdx.x;          // < 2*2304*1024
    int c = (int)(i & 1023);
    long rr = i >> 10;
    int b = (int)(rr / 2304);
    int r = (int)(rr - (long)b * 2304);
    float v;
    if (r < 256)       v = cmem[((long)b * 256  + r)        * 1024 + c];
    else if (r < 1280) v = mem [((long)b * 1024 + (r - 256))* 1024 + c];
    else               v = x   [((long)b * 1024 + (r - 1280))*1024 + c];
    dst[i] = (__bf16)v;
}

// wcb[o][kappa] = conv_w[o][i][r], kappa = r*1024 + i;  conv_w flat (o,i,r) r fastest
__global__ void pack_convw(const float* __restrict__ cw, __bf16* __restrict__ dst) {
    long i = (long)blockIdx.x * 256 + threadIdx.x;          // < 1024*4096
    int o   = (int)(i >> 12);
    int kap = (int)(i & 4095);
    int r   = kap >> 10;
    int ii  = kap & 1023;
    dst[i] = (__bf16)cw[((long)o * 1024 + ii) * 4 + r];
}

__global__ void aux_zero(float* p) { if (threadIdx.x == 0) p[0] = 0.0f; }

// ---------------------------------------------------------------------------
// 128x64-tile bf16 WMMA GEMM, software pipelined (loads cluster ahead of
// 8 back-to-back WMMAs with partial loadcnt waits).
// ---------------------------------------------------------------------------
template <int MODE>
__device__ __forceinline__ long a_offset(int gr) {
    if (MODE == 0) {            // x rows live at kvinb[b][1280 + s]
        int b = gr >> 10, s = gr & 1023;
        return ((long)b * 2304 + 1280 + s) * 1024;
    } else if (MODE == 1 || MODE == 2) {
        return (long)gr * 1024;
    } else {                    // mem rows: kvinb[b][256 + 4t], 4 rows contiguous = K 4096
        int b = gr >> 8, t = gr & 255;
        return ((long)b * 2304 + 256 + t * 4) * 1024;
    }
}

template <int MODE>
__global__ __launch_bounds__(128)
void gemm128x64(const __bf16* __restrict__ A, const __bf16* __restrict__ Wt, int K,
                float* __restrict__ outF, __bf16* __restrict__ outQ,
                __bf16* __restrict__ outK, __bf16* __restrict__ outV,
                const float* __restrict__ bias) {
    const int lane = threadIdx.x & 31;
    const int wv   = threadIdx.x >> 5;
    const int half = lane >> 4;
    const int l16  = lane & 15;
    const int row0 = blockIdx.x * 128 + wv * 32;
    const int n0   = blockIdx.y * 64;

    const __bf16* ar0 = A + a_offset<MODE>(row0 + l16);
    const __bf16* ar1 = A + a_offset<MODE>(row0 + 16 + l16);
    const __bf16* wb  = Wt + (long)(n0 + l16) * K + half * 16;

    const f32x8 zf = {0.f,0.f,0.f,0.f,0.f,0.f,0.f,0.f};
    f32x8 c0[4], c1[4];
    #pragma unroll
    for (int nt = 0; nt < 4; ++nt) { c0[nt] = zf; c1[nt] = zf; }

    bf16x16 a0c = load_a_op(ar0 + half * 16, ar0 + 8 + half * 16);
    bf16x16 a1c = load_a_op(ar1 + half * 16, ar1 + 8 + half * 16);
    bf16x16 bc[4];
    #pragma unroll
    for (int nt = 0; nt < 4; ++nt) bc[nt] = *(const bf16x16*)(wb + (long)nt * 16 * K);

    for (int k0 = 0; k0 < K; k0 += 32) {
        const int kn = (k0 + 32 < K) ? (k0 + 32) : k0;   // last iter: harmless reload
        bf16x16 a0n = load_a_op(ar0 + kn + half * 16, ar0 + kn + 8 + half * 16);
        bf16x16 a1n = load_a_op(ar1 + kn + half * 16, ar1 + kn + 8 + half * 16);
        bf16x16 bn[4];
        #pragma unroll
        for (int nt = 0; nt < 4; ++nt) bn[nt] = *(const bf16x16*)(wb + (long)nt * 16 * K + kn);

        #pragma unroll
        for (int nt = 0; nt < 4; ++nt) c0[nt] = wmma_bf16(a0c, bc[nt], c0[nt]);
        #pragma unroll
        for (int nt = 0; nt < 4; ++nt) c1[nt] = wmma_bf16(a1c, bc[nt], c1[nt]);

        a0c = a0n; a1c = a1n;
        #pragma unroll
        for (int nt = 0; nt < 4; ++nt) bc[nt] = bn[nt];
    }

    #pragma unroll
    for (int st = 0; st < 2; ++st) {
        #pragma unroll
        for (int nt = 0; nt < 4; ++nt) {
            #pragma unroll
            for (int g = 0; g < 8; ++g) {
                float val = (st == 0) ? c0[nt][g] : c1[nt][g];
                int grow = row0 + st * 16 + g + half * 8;   // C row = g + 8*half
                int col  = n0 + nt * 16 + l16;
                if (MODE == 2 || MODE == 3) {
                    outF[(long)grow * 1024 + col] = val + bias[col];
                } else if (MODE == 0) {
                    int b = grow >> 10, s = grow & 1023, h = col >> 6, d = col & 63;
                    outQ[(((long)(b * 16 + h)) * 1024 + s) * 64 + d] = (__bf16)val;
                } else {
                    int b = (grow >= 2304) ? 1 : 0;
                    int r = grow - b * 2304;
                    if (col < 1024) {
                        int h = col >> 6, d = col & 63;
                        outK[(((long)(b * 16 + h)) * 2304 + r) * 64 + d] = (__bf16)val;
                    } else {
                        int jj = col - 1024, h = jj >> 6, d = jj & 63;
                        outV[(((long)(b * 16 + h)) * 64 + d) * 2304 + r] = (__bf16)val;
                    }
                }
            }
        }
    }
}

// ---------------------------------------------------------------------------
// Fused flash attention with relative-position shift.
// grid (B*16, S/64); 128 threads = 4 waves; wave w owns query rows qr0..qr0+15.
// att[s,m] = scale*(q[s].k[m] + q[s].pos[1023+m-s]),  mask m <= s+1280.
// K/V tiles are shared by all 4 waves -> staged once per block in LDS with
// async global->LDS copies, double buffered (copy of tile kt+1 overlaps the
// WMMAs of tile kt).  ldsR/ldsP are per-wave: wave_barrier is enough there.
// ---------------------------------------------------------------------------
__global__ __launch_bounds__(128)
void attn_kernel(const __bf16* __restrict__ qb, const __bf16* __restrict__ kb,
                 const __bf16* __restrict__ vtb, const __bf16* __restrict__ posb,
                 __bf16* __restrict__ attout) {
    constexpr float scale = 0.125f;                 // 1/sqrt(64)
    const int tid  = threadIdx.x;
    const int lane = tid & 31;
    const int wv   = tid >> 5;
    const int half = lane >> 4;
    const int l16  = lane & 15;
    const int bh   = blockIdx.x;                    // b*16 + h
    const int h    = bh & 15;
    const int b    = bh >> 4;
    const int s0   = blockIdx.y * 64;
    const int qr0  = s0 + wv * 16;

    __shared__ __align__(64) __bf16 ldsK[2][64 * 72];   // padded rows: bank-conflict free
    __shared__ __align__(64) __bf16 ldsV[2][64 * 72];
    __shared__ __align__(64) float  ldsR[4][16][96];    // per-wave pos-logit window
    __shared__ __align__(64) __bf16 ldsP[4][16 * 72];   // per-wave P tile (A relayout)

    const __bf16* qrp = qb + ((long)bh * 1024 + qr0 + l16) * 64;
    bf16x16 qa0 = load_a_op(qrp + half * 16,      qrp + 8 + half * 16);
    bf16x16 qa1 = load_a_op(qrp + 32 + half * 16, qrp + 40 + half * 16);

    const __bf16* kbase = kb  + (long)bh * 2304 * 64;
    const __bf16* vbase = vtb + (long)bh * 64 * 2304;
    const __bf16* pbase = posb + (long)h * 2304 * 64;

    const f32x8 zf = {0.f,0.f,0.f,0.f,0.f,0.f,0.f,0.f};
    f32x8 acc[4];
    float run_m[8], run_l[8];
    #pragma unroll
    for (int nt = 0; nt < 4; ++nt) acc[nt] = zf;
    #pragma unroll
    for (int g = 0; g < 8; ++g) { run_m[g] = -3.0e38f; run_l[g] = 0.0f; }

    const int nkt = ((s0 + 1343) >> 6) + 1 < 36 ? ((s0 + 1343) >> 6) + 1 : 36;

    // prologue: stage tile 0
    stage_tile(ldsK[0], kbase, 64, tid);
    stage_tile(ldsV[0], vbase, 2304, tid);

    for (int kt = 0; kt < nkt; ++kt) {
        const int m0  = kt * 64;
        const int buf = kt & 1;

        __syncthreads();                    // everyone done reading buf^1 (prev tile)
        if (kt + 1 < nkt) {                 // async-stage next tile, overlap with compute
            stage_tile(ldsK[buf ^ 1], kbase + (long)(m0 + 64) * 64, 64, tid);
            stage_tile(ldsV[buf ^ 1], vbase + (m0 + 64), 2304, tid);
            ASYNC_WAIT(8);                  // 8 newest in flight; tile kt complete
        } else {
            ASYNC_WAIT(0);
        }
        __syncthreads();                    // tile kt visible to all waves

        const __bf16* ldk = ldsK[buf];
        const __bf16* ldv = ldsV[buf];

        // ---- scores S = q . k^T (16x64), B operands from LDS ----
        f32x8 sc[4];
        #pragma unroll
        for (int nt = 0; nt < 4; ++nt) {
            const __bf16* kr = ldk + (nt * 16 + l16) * 72;
            sc[nt] = wmma_bf16(qa0, *(const bf16x16*)(kr + half * 16), zf);
            sc[nt] = wmma_bf16(qa1, *(const bf16x16*)(kr + 32 + half * 16), sc[nt]);
        }

        // ---- pos window R[s,p] -> LDS, pipelined global loads ----
        {
            const int pw0 = 1008 + m0 - qr0;        // = 1023 + m0 - (qr0+15)
            int p = pw0 + l16;
            p = p < 0 ? 0 : (p > 2303 ? 2303 : p);
            const __bf16* pr = pbase + (long)p * 64;
            bf16x16 b0 = *(const bf16x16*)(pr + half * 16);
            bf16x16 b1 = *(const bf16x16*)(pr + 32 + half * 16);
            #pragma unroll
            for (int nt2 = 0; nt2 < 6; ++nt2) {
                const int ntn = nt2 < 5 ? nt2 + 1 : nt2;
                int pn = pw0 + ntn * 16 + l16;
                pn = pn < 0 ? 0 : (pn > 2303 ? 2303 : pn);
                const __bf16* prn = pbase + (long)pn * 64;
                bf16x16 nb0 = *(const bf16x16*)(prn + half * 16);
                bf16x16 nb1 = *(const bf16x16*)(prn + 32 + half * 16);
                f32x8 r = wmma_bf16(qa0, b0, zf);
                r = wmma_bf16(qa1, b1, r);
                #pragma unroll
                for (int g = 0; g < 8; ++g)
                    ldsR[wv][g + half * 8][nt2 * 16 + l16] = r[g];
                b0 = nb0; b1 = nb1;
            }
        }
        __builtin_amdgcn_wave_barrier();

        // ---- gather shifted pos logits, mask, online softmax (P in-place in sc) ----
        float tmax[8];
        #pragma unroll
        for (int g = 0; g < 8; ++g) tmax[g] = -3.0e38f;
        #pragma unroll
        for (int nt = 0; nt < 4; ++nt) {
            #pragma unroll
            for (int g = 0; g < 8; ++g) {
                int s_l = g + half * 8;
                int s   = qr0 + s_l;
                int m   = m0 + nt * 16 + l16;
                float v = scale * (sc[nt][g] + ldsR[wv][s_l][nt * 16 + l16 + 15 - s_l]);
                v = (m <= s + 1280) ? v : -3.0e38f;
                sc[nt][g] = v;
                tmax[g] = fmaxf(tmax[g], v);
            }
        }
        #pragma unroll
        for (int g = 0; g < 8; ++g) tmax[g] = halfmax16(tmax[g]);

        float corr[8];
        #pragma unroll
        for (int g = 0; g < 8; ++g) {
            float nm = fmaxf(run_m[g], tmax[g]);
            corr[g] = __expf(run_m[g] - nm);
            run_m[g] = nm;
        }
        float rsum[8];
        #pragma unroll
        for (int g = 0; g < 8; ++g) rsum[g] = 0.0f;
        #pragma unroll
        for (int nt = 0; nt < 4; ++nt)
            #pragma unroll
            for (int g = 0; g < 8; ++g) {
                float e = __expf(sc[nt][g] - run_m[g]);
                sc[nt][g] = e;
                rsum[g] += e;
            }
        #pragma unroll
        for (int g = 0; g < 8; ++g) {
            rsum[g] = halfsum16(rsum[g]);
            run_l[g] = run_l[g] * corr[g] + rsum[g];
        }
        #pragma unroll
        for (int nt = 0; nt < 4; ++nt)
            #pragma unroll
            for (int g = 0; g < 8; ++g) acc[nt][g] *= corr[g];

        // ---- P to LDS (bf16), reload in WMMA-A layout, acc += P @ V ----
        #pragma unroll
        for (int nt = 0; nt < 4; ++nt)
            #pragma unroll
            for (int g = 0; g < 8; ++g)
                ldsP[wv][(g + half * 8) * 72 + nt * 16 + l16] = (__bf16)sc[nt][g];
        __builtin_amdgcn_wave_barrier();

        {
            const __bf16* prow = &ldsP[wv][l16 * 72];
            bf16x16 pa0 = load_a_op(prow + half * 16,      prow + 8 + half * 16);
            bf16x16 pa1 = load_a_op(prow + 32 + half * 16, prow + 40 + half * 16);
            #pragma unroll
            for (int nt = 0; nt < 4; ++nt) {
                const __bf16* vr = ldv + (nt * 16 + l16) * 72;
                acc[nt] = wmma_bf16(pa0, *(const bf16x16*)(vr + half * 16), acc[nt]);
                acc[nt] = wmma_bf16(pa1, *(const bf16x16*)(vr + 32 + half * 16), acc[nt]);
            }
        }
        __builtin_amdgcn_wave_barrier();
    }

    // ---- epilogue: normalize, write attout (B,S,H) bf16 ----
    #pragma unroll
    for (int nt = 0; nt < 4; ++nt) {
        #pragma unroll
        for (int g = 0; g < 8; ++g) {
            int s_l = g + half * 8;
            int s   = qr0 + s_l;
            float o = acc[nt][g] / run_l[g];
            attout[((long)b * 1024 + s) * 1024 + h * 64 + nt * 16 + l16] = (__bf16)o;
        }
    }
}

// ---------------------------------------------------------------------------
extern "C" void kernel_launch(void* const* d_in, const int* in_sizes, int n_in,
                              void* d_out, int out_size, void* d_ws, size_t ws_size,
                              hipStream_t stream) {
    const float* x       = (const float*)d_in[0];
    const float* mem     = (const float*)d_in[1];
    const float* cmem    = (const float*)d_in[2];
    const float* posf    = (const float*)d_in[3];
    const float* W_q     = (const float*)d_in[4];
    const float* W_kv    = (const float*)d_in[5];
    const float* W_proj  = (const float*)d_in[6];
    const float* b_proj  = (const float*)d_in[7];
    const float* conv_w  = (const float*)d_in[8];
    const float* conv_b  = (const float*)d_in[9];
    float* out = (float*)d_out;

    char* ws = (char*)d_ws;
    __bf16* kvinb   = (__bf16*)(ws + 0);           // 2*2304*1024          (9,437,184 B)
    __bf16* wqb     = (__bf16*)(ws + 9437184);     // 1024*1024            (2,097,152 B)
    __bf16* wkvb    = (__bf16*)(ws + 11534336);    // 2048*1024            (4,194,304 B)
    __bf16* wpb     = (__bf16*)(ws + 15728640);    // 1024*1024            (2,097,152 B)
    __bf16* wcb     = (__bf16*)(ws + 17825792);    // 1024*4096            (8,388,608 B)
    __bf16* posb    = (__bf16*)(ws + 26214400);    // 16*2304*64           (4,718,592 B)
    __bf16* qb      = (__bf16*)(ws + 30932992);    // 32*1024*64           (4,194,304 B)
    __bf16* kb      = (__bf16*)(ws + 35127296);    // 32*2304*64           (9,437,184 B)
    __bf16* vtb     = (__bf16*)(ws + 44564480);    // 32*64*2304           (9,437,184 B)
    __bf16* attoutb = (__bf16*)(ws + 54001664);    // 2*1024*1024          (4,194,304 B)
    // total 58,195,968 B

    // ---- pack to bf16 ----
    pack_f32_bf16<<<4096,  256, 0, stream>>>(W_q,    wqb,  1048576);
    pack_f32_bf16<<<8192,  256, 0, stream>>>(W_kv,   wkvb, 2097152);
    pack_f32_bf16<<<4096,  256, 0, stream>>>(W_proj, wpb,  1048576);
    pack_f32_bf16<<<9216,  256, 0, stream>>>(posf,   posb, 2359296);
    pack_kvin    <<<18432, 256, 0, stream>>>(x, mem, cmem, kvinb);
    pack_convw   <<<16384, 256, 0, stream>>>(conv_w, wcb);

    // ---- projections (bf16 WMMA) ----
    gemm128x64<0><<<dim3(16, 16), 128, 0, stream>>>(kvinb, wqb, 1024, nullptr, qb, nullptr, nullptr, nullptr);
    gemm128x64<1><<<dim3(36, 32), 128, 0, stream>>>(kvinb, wkvb, 1024, nullptr, nullptr, kb, vtb, nullptr);

    // ---- fused flash attention with relative-position shift ----
    attn_kernel<<<dim3(32, 16), 128, 0, stream>>>(qb, kb, vtb, posb, attoutb);

    // ---- output projection: out = attout @ W_proj^T + b_proj ----
    gemm128x64<2><<<dim3(16, 16), 128, 0, stream>>>(attoutb, wpb, 1024, out, nullptr, nullptr, nullptr, b_proj);

    // ---- new_mem = x (d2d copy) ----
    (void)hipMemcpyAsync(out + 2097152, x, 2097152 * sizeof(float), hipMemcpyDeviceToDevice, stream);

    // ---- new_cmem = conv-compress(mem): GEMM, M=512, K=4096 ----
    gemm128x64<3><<<dim3(4, 16), 128, 0, stream>>>(kvinb, wcb, 4096, out + 4194304, nullptr, nullptr, nullptr, conv_b);

    // ---- aux loss ----
    aux_zero<<<1, 1, 0, stream>>>(out + 4718592);
}
